// CoxPartialLikelihoodLoss_76974403879428
// MI455X (gfx1250) — matrix operand order repaired and checked by reading, not compiled, not run on hardware
//
#include <hip/hip_runtime.h>

typedef _Float16 v16h __attribute__((ext_vector_type(16)));
typedef _Float16 v8h  __attribute__((ext_vector_type(8)));
typedef float    v8f  __attribute__((ext_vector_type(8)));
typedef float    v4f  __attribute__((ext_vector_type(4)));

// ---------------------------------------------------------------------------
// Kernel 1: split labels, compute exp(theta) in fp16 (B-matrix operand) and
// keep event times in f32 so the <= comparisons match the reference exactly.
// ---------------------------------------------------------------------------
__global__ __launch_bounds__(256) void cox_prep(const float* __restrict__ theta,
                                                const float* __restrict__ ylab,
                                                float* __restrict__ t,
                                                _Float16* __restrict__ e16,
                                                int n) {
    int i = blockIdx.x * blockDim.x + threadIdx.x;
    if (i < n) {
        t[i]   = ylab[2 * i];                       // y_time
        e16[i] = (_Float16)expf(theta[i]);          // exp(theta)
    }
}

// ---------------------------------------------------------------------------
// Kernel 2: denom = risk_mat @ exp_theta via v_wmma_f32_16x16x32_f16.
// One wave owns a 16-row tile; A = 16x32 fp16 risk indicators built with f32
// compares; B = exp(theta) broadcast across all 16 columns; f32 accumulate.
// ---------------------------------------------------------------------------
__global__ __launch_bounds__(256) void cox_denom_wmma(const float* __restrict__ t,
                                                      const _Float16* __restrict__ e16,
                                                      float* __restrict__ denom,
                                                      int n) {
    const int lane = threadIdx.x & 31;
    const int wave = (blockIdx.x * blockDim.x + threadIdx.x) >> 5;   // row tile
    const int col  = lane & 15;          // output column / A row
    const int hs   = lane >> 4;          // half-wave select
    const int rowbase = wave * 16;

    const float trow = t[rowbase + col]; // t_i for this lane's A row
    const int   ko   = hs * 8;           // A K-offset per ISA layout

    const _Float16 h1 = (_Float16)1.0f;
    const _Float16 h0 = (_Float16)0.0f;

    v8f acc = {};
    for (int jbase = 0; jbase < n; jbase += 32) {
        // A-operand t values: K = {ko..ko+7, ko+16..ko+23}  (all 16B aligned)
        v4f ta0 = *(const v4f*)(t + jbase + ko);
        v4f ta1 = *(const v4f*)(t + jbase + ko + 4);
        v4f ta2 = *(const v4f*)(t + jbase + ko + 16);
        v4f ta3 = *(const v4f*)(t + jbase + ko + 20);
        // B-operand exp values: K = {8*hs..8*hs+7, 8*hs+16..8*hs+23}
        v8h eb0 = *(const v8h*)(e16 + jbase + hs * 8);
        v8h eb1 = *(const v8h*)(e16 + jbase + hs * 8 + 16);

        v16h a, b;
#pragma unroll
        for (int h = 0; h < 4; ++h) {
            a[h]      = (ta0[h] <= trow) ? h1 : h0;
            a[h + 4]  = (ta1[h] <= trow) ? h1 : h0;
            a[h + 8]  = (ta2[h] <= trow) ? h1 : h0;
            a[h + 12] = (ta3[h] <= trow) ? h1 : h0;
        }
#pragma unroll
        for (int h = 0; h < 8; ++h) {
            b[h]     = eb0[h];
            b[h + 8] = eb1[h];
        }

        // D = A x B + C : every output column n gets the same partial denom
        acc = __builtin_amdgcn_wmma_f32_16x16x32_f16(
            /*neg_a=*/false, a, /*neg_b=*/false, b,
            /*c_mod=*/(short)0, acc, /*reuse_a=*/false, /*reuse_b=*/false);
    }

    // D layout: lane L (col==0) holds M = hs*8 + r in acc[r]
    if (col == 0) {
#pragma unroll
        for (int r = 0; r < 8; ++r)
            denom[rowbase + hs * 8 + r] = acc[r];
    }
}

// ---------------------------------------------------------------------------
// Kernel 3: nll_i = -(theta_i - log(denom_i + 1e-9)) * event_i; mean over N.
// Single-block deterministic tree reduction.
// ---------------------------------------------------------------------------
__global__ __launch_bounds__(256) void cox_reduce(const float* __restrict__ theta,
                                                  const float* __restrict__ ylab,
                                                  const float* __restrict__ denom,
                                                  float* __restrict__ out,
                                                  int n) {
    __shared__ float s[256];
    const int tid = threadIdx.x;
    float acc = 0.0f;
    for (int i = tid; i < n; i += 256) {
        float lc  = logf(denom[i] + 1e-9f);
        acc += -(theta[i] - lc) * ylab[2 * i + 1];
    }
    s[tid] = acc;
    __syncthreads();
#pragma unroll
    for (int off = 128; off > 0; off >>= 1) {
        if (tid < off) s[tid] += s[tid + off];
        __syncthreads();
    }
    if (tid == 0) out[0] = s[0] / (float)n;
}

// ---------------------------------------------------------------------------
extern "C" void kernel_launch(void* const* d_in, const int* in_sizes, int n_in,
                              void* d_out, int out_size, void* d_ws, size_t ws_size,
                              hipStream_t stream) {
    const float* theta = (const float*)d_in[0];   // (N,) f32
    const float* ylab  = (const float*)d_in[1];   // (N,2) f32 [time, event]
    float*       out   = (float*)d_out;           // scalar f32
    const int n = in_sizes[0];                    // 16384

    // workspace layout: t[N] f32 | e16[N] f16 | denom[N] f32  (10*N bytes)
    float*    t     = (float*)d_ws;
    _Float16* e16   = (_Float16*)(t + n);
    float*    denom = (float*)((char*)d_ws + (size_t)n * 6);

    cox_prep<<<(n + 255) / 256, 256, 0, stream>>>(theta, ylab, t, e16, n);

    const int waves  = n / 16;      // one wave per 16-row tile -> 1024 waves
    const int blocks = waves / 8;   // 8 waves (256 threads) per block
    cox_denom_wmma<<<blocks, 256, 0, stream>>>(t, e16, denom, n);

    cox_reduce<<<1, 256, 0, stream>>>(theta, ylab, denom, out, n);
}
